// NCDE_24318104830317
// MI455X (gfx1250) — matrix operand order repaired and checked by reading, not compiled, not run on hardware
//
#include <hip/hip_runtime.h>

// ---------------- CDNA5 WMMA types ----------------
typedef __bf16 bf16_t;
typedef __attribute__((ext_vector_type(16))) __bf16 v16bf;
typedef __attribute__((ext_vector_type(8)))  float  v8f;

#define DT    0.1f
#define LSEQ  512
#define XDIM  16
#define ZDIM  64
#define HDIM  128

// D = A(16x32 bf16) * B(32x16 bf16) + C(16x16 f32)
__device__ __forceinline__ v8f wmma_bf16(v16bf a, v16bf b, v8f c) {
  return __builtin_amdgcn_wmma_f32_16x16x32_bf16(false, a, false, b, (short)0, c,
                                                 false, false);
}

// Gather an A-fragment (16x32, bf16) from a row-major f32 activation tile
// act[16][stride].  ISA layout: lane l holds row m=l&15; element e maps to
// k = kb*32 + (e<8?e:e+8) + (l<16?0:8).  Pairs (e=2r,2r+1) are k-consecutive,
// so this lowers to ds_load_b128 + v_cvt_pk_bf16_f32.
__device__ __forceinline__ v16bf gatherA(const float* act, int stride, int kb,
                                         int lane, int kmax) {
  v16bf a;
  const int m    = lane & 15;
  const int kofs = (lane < 16) ? 0 : 8;
#pragma unroll
  for (int e = 0; e < 16; ++e) {
    int   k = kb * 32 + (e < 8 ? e : e + 8) + kofs;
    float f = (k < kmax) ? act[m * stride + k] : 0.0f;
    a[e]    = (bf16_t)f;
  }
  return a;
}

// Load a pre-staged B-fragment: layout [tile][kstep][lane][16 bf16] (32B aligned).
__device__ __forceinline__ v16bf loadBfrag(const bf16_t* base, int tile,
                                           int ksteps, int kb, int lane) {
  return *(const v16bf*)(base + ((size_t)(tile * ksteps + kb) * 32 + lane) * 16);
}

// ---------------- prepass: f32 weights -> bf16 B-fragment layout ----------------
// B-fragment element e of lane l: k = kb*32 + (e<8?e:e+8) + (l<16?0:8), n = tile*16 + (l&15)
__global__ void conv_bfrag(const float* __restrict__ w, bf16_t* __restrict__ outp,
                           int K, int N, int ksteps, int total) {
  int idx = blockIdx.x * blockDim.x + threadIdx.x;
  if (idx >= total) return;
  int e    = idx & 15;
  int lane = (idx >> 4) & 31;
  int g    = idx >> 9;
  int kb   = g % ksteps;
  int tile = g / ksteps;
  int k    = kb * 32 + (e < 8 ? e : e + 8) + ((lane < 16) ? 0 : 8);
  int n    = tile * 16 + (lane & 15);
  float v  = (k < K) ? w[(size_t)k * N + n] : 0.0f;
  outp[idx] = (bf16_t)v;
}

// ---------------- persistent scan kernel: 8 blocks x 256 threads ----------------
__global__ __launch_bounds__(256) void ncde_scan(
    const float* __restrict__ x,
    const float* __restrict__ dyn_b1, const float* __restrict__ dyn_b2,
    const float* __restrict__ dyn_b3,
    const float* __restrict__ ini_b1, const float* __restrict__ ini_b2,
    const float* __restrict__ ini_b3,
    const bf16_t* __restrict__ dynW1f, const bf16_t* __restrict__ dynW2f,
    const bf16_t* __restrict__ dynW3f,
    const bf16_t* __restrict__ iniW1f, const bf16_t* __restrict__ iniW2f,
    const bf16_t* __restrict__ iniW3f,
    float* __restrict__ out) {
  __shared__ float zcur[16][64];    // current latent state (f32, full precision)
  __shared__ float zin[16][64];     // RK4 stage input
  __shared__ float hb1[16][128];
  __shared__ float hb2[16][128];
  __shared__ float rhsS[16][64];    // stage rhs (k_s)
  __shared__ float kacc[16][64];    // k1 + 2k2 + 2k3 (+ k4)
  __shared__ float vtile[16][16];   // dX/dt for this interval
  __shared__ float vsum[16];        // sum_x v[b,x] (for -0.001*z term)
  __shared__ float b3s[ZDIM * XDIM];                 // dyn_b3 (indexed by zt)
  __shared__ __align__(32) bf16_t w1f[8 * 2 * 512];  // dyn_w1 fragments in LDS

  const int tid  = threadIdx.x;
  const int wave = tid >> 5;
  const int lane = tid & 31;
  const int bb   = blockIdx.x * 16;         // batch-row base (8 blocks, B=128)
  const int nlo  = lane & 15;               // n within tile
  const int hsel = (lane < 16) ? 0 : 8;     // C/D row offset per lane half

  // stage dyn_w1 fragments + dyn_b3 into LDS
  for (int i = tid; i < 8 * 2 * 512; i += 256) w1f[i] = dynW1f[i];
  for (int i = tid; i < ZDIM * XDIM; i += 256) b3s[i] = dyn_b3[i];

  // hoist scalar bias loads out of the scan
  const float b1v  = dyn_b1[wave * 16 + nlo];
  const float b2v  = dyn_b2[wave * 16 + nlo];
  const float ib1v = ini_b1[wave * 16 + nlo];
  const float ib2v = ini_b2[wave * 16 + nlo];
  const float ib3v = (wave < 4) ? ini_b3[wave * 16 + nlo] : 0.0f;

  // ================= init MLP: z0 = ini(x[:, 0]) =================
  {
    int m = tid >> 4, xx = tid & 15;
    vtile[m][xx] = x[((size_t)(bb + m) * LSEQ + 0) * XDIM + xx];
  }
  __syncthreads();
  {  // L1: [16,16(pad to 32)] @ [16,128]
    v8f   acc = {};
    v16bf a   = gatherA(&vtile[0][0], 16, 0, lane, 16);
    v16bf b   = loadBfrag(iniW1f, wave, 1, 0, lane);
    acc       = wmma_bf16(a, b, acc);
#pragma unroll
    for (int r = 0; r < 8; ++r) {
      float v = acc[r] + ib1v;
      hb1[r + hsel][wave * 16 + nlo] = v > 0.0f ? v : 0.0f;
    }
  }
  __syncthreads();
  {  // L2: [16,128] @ [128,128]
    v8f acc = {};
#pragma unroll
    for (int kb = 0; kb < 4; ++kb) {
      v16bf a = gatherA(&hb1[0][0], 128, kb, lane, 128);
      v16bf b = loadBfrag(iniW2f, wave, 4, kb, lane);
      acc     = wmma_bf16(a, b, acc);
    }
#pragma unroll
    for (int r = 0; r < 8; ++r) {
      float v = acc[r] + ib2v;
      hb2[r + hsel][wave * 16 + nlo] = v > 0.0f ? v : 0.0f;
    }
  }
  __syncthreads();
  if (wave < 4) {  // L3: [16,128] @ [128,64] -> z0 (wave-uniform branch, EXEC full)
    v8f acc = {};
#pragma unroll
    for (int kb = 0; kb < 4; ++kb) {
      v16bf a = gatherA(&hb2[0][0], 128, kb, lane, 128);
      v16bf b = loadBfrag(iniW3f, wave, 4, kb, lane);
      acc     = wmma_bf16(a, b, acc);
    }
#pragma unroll
    for (int r = 0; r < 8; ++r) zcur[r + hsel][wave * 16 + nlo] = acc[r] + ib3v;
  }
  __syncthreads();
  for (int idx = tid; idx < 16 * 64; idx += 256) {  // emit z at l = 0
    int m = idx >> 6, z = idx & 63;
    out[((size_t)(bb + m) * LSEQ + 0) * ZDIM + z] = zcur[m][z];
  }

  // ================= RK4 scan: 511 steps (z_1 .. z_{L-1}) =================
  // Loops kept ROLLED (#pragma unroll 1) so the hot loop fits in I$ and VGPR
  // use stays below 256 (avoids s_set_vgpr_msb thrash seen when fully unrolled).
  const float inv_dt = 1.0f / DT;
#pragma unroll 1
  for (int i = 0; i < LSEQ - 1; ++i) {
    {  // piecewise-constant dX/dt on this interval
      int   m = tid >> 4, xx = tid & 15;
      float x0v = x[((size_t)(bb + m) * LSEQ + i) * XDIM + xx];
      float x1v = x[((size_t)(bb + m) * LSEQ + i + 1) * XDIM + xx];
      vtile[m][xx] = (x1v - x0v) * inv_dt;
    }
    __syncthreads();
    if (tid < 16) {
      float s = 0.0f;
#pragma unroll
      for (int xx = 0; xx < 16; ++xx) s += vtile[tid][xx];
      vsum[tid] = s;
    }
    __syncthreads();
    float vv[8];  // v[b=r+hsel, x=nlo] per accumulator register
#pragma unroll
    for (int r = 0; r < 8; ++r) vv[r] = vtile[r + hsel][nlo];

#pragma unroll 1
    for (int s = 0; s < 4; ++s) {
      const float* act = (s == 0) ? &zcur[0][0] : &zin[0][0];
      {  // L1: [16,64] @ [64,128], B-frags in LDS
        v8f acc = {};
#pragma unroll
        for (int kb = 0; kb < 2; ++kb) {
          v16bf a = gatherA(act, 64, kb, lane, 64);
          v16bf b = loadBfrag(w1f, wave, 2, kb, lane);
          acc     = wmma_bf16(a, b, acc);
        }
#pragma unroll
        for (int r = 0; r < 8; ++r) {
          float v = acc[r] + b1v;
          hb1[r + hsel][wave * 16 + nlo] = v > 0.0f ? v : 0.0f;
        }
      }
      __syncthreads();
      {  // L2: [16,128] @ [128,128], B-frags stream from L2 cache
        v8f acc = {};
#pragma unroll
        for (int kb = 0; kb < 4; ++kb) {
          v16bf a = gatherA(&hb1[0][0], 128, kb, lane, 128);
          v16bf b = loadBfrag(dynW2f, wave, 4, kb, lane);
          acc     = wmma_bf16(a, b, acc);
        }
#pragma unroll
        for (int r = 0; r < 8; ++r) {
          float v = acc[r] + b2v;
          hb2[r + hsel][wave * 16 + nlo] = v > 0.0f ? v : 0.0f;
        }
      }
      __syncthreads();
      {  // L3: [16,128] @ [128,1024] + tanh, fused contraction over x
        v16bf a3[4];
#pragma unroll
        for (int kb = 0; kb < 4; ++kb)
          a3[kb] = gatherA(&hb2[0][0], 128, kb, lane, 128);
#pragma unroll 1
        for (int t8 = 0; t8 < 8; ++t8) {
          int zt  = wave * 8 + t8;  // one 16-col tile == one z, all 16 x
          // prefetch next tile's w3 fragments into cache (global_prefetch_b8)
          int ztn = (t8 < 7) ? zt + 1 : zt;
#pragma unroll
          for (int kb = 0; kb < 4; ++kb)
            __builtin_prefetch(
                dynW3f + ((size_t)(ztn * 4 + kb) * 32 + lane) * 16, 0, 1);
          v8f acc = {};
#pragma unroll
          for (int kb = 0; kb < 4; ++kb) {
            v16bf b = loadBfrag(dynW3f, zt, 4, kb, lane);
            acc     = wmma_bf16(a3[kb], b, acc);
          }
          float b3v = b3s[zt * 16 + nlo];
#pragma unroll
          for (int r = 0; r < 8; ++r) {
            float g = tanhf(acc[r] + b3v);
            float p = g * vv[r];  // g[b, z, x=nlo] * v[b, x=nlo]
            p += __shfl_xor(p, 1);
            p += __shfl_xor(p, 2);
            p += __shfl_xor(p, 4);
            p += __shfl_xor(p, 8);  // sum over x within each 16-lane half
            if (nlo == 0) rhsS[r + hsel][zt] = p;
          }
        }
      }
      __syncthreads();
      // RK4 bookkeeping (adds -0.001*z*sum_x(v) damping term here)
      for (int idx = tid; idx < 16 * 64; idx += 256) {
        int   m  = idx >> 6, z = idx & 63;
        float rv = rhsS[m][z] - 0.001f * act[idx] * vsum[m];
        if (s == 0) {
          kacc[m][z] = rv;
          zin[m][z]  = zcur[m][z] + 0.5f * DT * rv;
        } else if (s == 1) {
          kacc[m][z] += 2.0f * rv;
          zin[m][z] = zcur[m][z] + 0.5f * DT * rv;
        } else if (s == 2) {
          kacc[m][z] += 2.0f * rv;
          zin[m][z] = zcur[m][z] + DT * rv;
        } else {
          float zn = zcur[m][z] + (DT / 6.0f) * (kacc[m][z] + rv);
          zcur[m][z] = zn;
          out[((size_t)(bb + m) * LSEQ + (i + 1)) * ZDIM + z] = zn;
        }
      }
      __syncthreads();
    }
  }
}

// ---------------- launch ----------------
extern "C" void kernel_launch(void* const* d_in, const int* in_sizes, int n_in,
                              void* d_out, int out_size, void* d_ws,
                              size_t ws_size, hipStream_t stream) {
  const float* x       = (const float*)d_in[1];
  const float* dyn_w1  = (const float*)d_in[2];
  const float* dyn_b1  = (const float*)d_in[3];
  const float* dyn_w2  = (const float*)d_in[4];
  const float* dyn_b2  = (const float*)d_in[5];
  const float* dyn_w3  = (const float*)d_in[6];
  const float* dyn_b3  = (const float*)d_in[7];
  const float* init_w1 = (const float*)d_in[8];
  const float* init_b1 = (const float*)d_in[9];
  const float* init_w2 = (const float*)d_in[10];
  const float* init_b2 = (const float*)d_in[11];
  const float* init_w3 = (const float*)d_in[12];
  const float* init_b3 = (const float*)d_in[13];

  // bf16 fragment workspace layout (elements)
  bf16_t* ws     = (bf16_t*)d_ws;
  bf16_t* dW1f   = ws;               // 8 tiles * 2 ksteps * 512 =   8192
  bf16_t* dW2f   = dW1f + 8192;      // 8 * 4 * 512              =  16384
  bf16_t* dW3f   = dW2f + 16384;     // 64 * 4 * 512             = 131072
  bf16_t* iW1f   = dW3f + 131072;    // 8 * 1 * 512              =   4096
  bf16_t* iW2f   = iW1f + 4096;      // 8 * 4 * 512              =  16384
  bf16_t* iW3f   = iW2f + 16384;     // 4 * 4 * 512              =   8192

  auto conv = [&](const float* w, bf16_t* o, int K, int N, int ksteps,
                  int ntiles) {
    int total = ntiles * ksteps * 512;
    conv_bfrag<<<(total + 255) / 256, 256, 0, stream>>>(w, o, K, N, ksteps,
                                                        total);
  };
  conv(dyn_w1, dW1f, 64, 128, 2, 8);
  conv(dyn_w2, dW2f, 128, 128, 4, 8);
  conv(dyn_w3, dW3f, 128, 1024, 4, 64);
  conv(init_w1, iW1f, 16, 128, 1, 8);
  conv(init_w2, iW2f, 128, 128, 4, 8);
  conv(init_w3, iW3f, 128, 64, 4, 4);

  ncde_scan<<<8, 256, 0, stream>>>(x, dyn_b1, dyn_b2, dyn_b3, init_b1, init_b2,
                                   init_b3, dW1f, dW2f, dW3f, iW1f, iW2f, iW3f,
                                   (float*)d_out);
}
